// CapsuleBase_9174050144872
// MI455X (gfx1250) — matrix operand order repaired and checked by reading, not compiled, not run on hardware
//
#include <hip/hip_runtime.h>
#include <math.h>

#define NN 50000
#define EE 500000
#define KK 3
#define DD 128
#define RR 50
#define KD  (KK * DD)       /* 384 */
#define EF  (EE + NN)       /* 550000 edges incl. self loops */
#define SELF_REL (2 * RR)   /* 100 */
#define NEG_SLOPE 0.2f
#define BN_EPS    1e-5f
#define NTILES_N  (KD / 16) /* 24 */

typedef __attribute__((ext_vector_type(16))) _Float16 v16h;
typedef __attribute__((ext_vector_type(8)))  _Float16 v8h;
typedef __attribute__((ext_vector_type(8)))  float    v8f;

// ---------------------------------------------------------------------------
// order-preserving float<->uint encoding for atomicMax-based segment max
// ---------------------------------------------------------------------------
__device__ __forceinline__ unsigned encOrdered(float x) {
    unsigned b = __float_as_uint(x);
    return (b & 0x80000000u) ? ~b : (b | 0x80000000u);
}
__device__ __forceinline__ float decOrdered(unsigned e) {
    unsigned b = (e & 0x80000000u) ? (e ^ 0x80000000u) : ~e;
    return __uint_as_float(b);
}
#define ENC_NEG_INF 0x007FFFFFu   /* encOrdered(-inf) */

// ---------------------------------------------------------------------------
// kernel 0: initialize workspace accumulators
// ---------------------------------------------------------------------------
__global__ void k_init(float* __restrict__ outagg, float* __restrict__ asum,
                       unsigned* __restrict__ amaxe, float* __restrict__ colsum,
                       float* __restrict__ colsumsq) {
    long long stride = (long long)gridDim.x * blockDim.x;
    long long gid = (long long)blockIdx.x * blockDim.x + threadIdx.x;
    for (long long i = gid; i < (long long)NN * KD; i += stride) outagg[i] = 0.0f;
    for (long long i = gid; i < (long long)NN * KK; i += stride) {
        asum[i] = 0.0f;
        amaxe[i] = ENC_NEG_INF;
    }
    for (long long i = gid; i < KD; i += stride) { colsum[i] = 0.0f; colsumsq[i] = 0.0f; }
}

// ---------------------------------------------------------------------------
// kernel P1: convert init_embed fp32 -> f16 row-major [NN,128]
// ---------------------------------------------------------------------------
__global__ __launch_bounds__(256)
void k_pack_a(const float* __restrict__ A, _Float16* __restrict__ Ah) {
    long long gid = (long long)blockIdx.x * blockDim.x + threadIdx.x;
    long long n4 = (long long)NN * DD / 4;
    if (gid >= n4) return;
    float4 v = ((const float4*)A)[gid];
    _Float16* o = Ah + gid * 4;
    o[0] = (_Float16)v.x; o[1] = (_Float16)v.y;
    o[2] = (_Float16)v.z; o[3] = (_Float16)v.w;
}

// ---------------------------------------------------------------------------
// kernel P2: pack pca_w fp32 [128,384] into WMMA-B-fragment f16 layout:
//   Wp[ kstep(4) ][ ntile(24) ][ lane(32) ][ slot(16) ]
//   lane = hi*16 + (n&15),  slot j -> K = kstep*32 + hi*16 + j
// ---------------------------------------------------------------------------
__global__ __launch_bounds__(256)
void k_pack_w(const float* __restrict__ W, _Float16* __restrict__ Wp) {
    int gid = blockIdx.x * blockDim.x + threadIdx.x;
    if (gid >= DD * KD) return;
    int k = gid / KD;          // 0..127
    int n = gid % KD;          // 0..383
    int ks   = k >> 5;         // kstep
    int kin  = k & 31;
    int hi   = kin >> 4;
    int slot = kin & 15;
    int nt   = n >> 4;
    int lane = hi * 16 + (n & 15);
    long long idx = (((long long)ks * NTILES_N + nt) * 32 + lane) * 16 + slot;
    Wp[idx] = (_Float16)W[gid];
}

// ---------------------------------------------------------------------------
// kernel 1: x = init_embed @ pca_w + pca_b   via WMMA f32<-f16
// one wave32 per 16x16 output tile; K = 128 in 4 steps of 32.
// A fragment: 2x contiguous 16B loads/lane; B fragment: one 32B load/lane.
// NN % 16 == 0, so the store epilogue is unguarded (no exec-mask churn).
// ---------------------------------------------------------------------------
__global__ __launch_bounds__(32)
void k_pca_wmma(const _Float16* __restrict__ Ah,  // [NN,128] f16
                const _Float16* __restrict__ Wp,  // packed B fragments
                const float* __restrict__ bias,   // [384]
                float* __restrict__ X) {          // [NN,384]
    const int tile = blockIdx.x;
    const int m0 = (tile / NTILES_N) * 16;
    const int nt = tile % NTILES_N;
    const int lane = threadIdx.x;              // 0..31 (wave32)
    const int hi   = lane >> 4;
    const int lm   = lane & 15;

    const _Float16* arow = Ah + (long long)(m0 + lm) * DD + hi * 16;
    const v16h* bfrag = (const v16h*)(Wp) + (long long)nt * 32 + lane;

    v8f acc = {};
#pragma unroll
    for (int ks = 0; ks < 4; ++ks) {
        // A: slot j -> K = ks*32 + hi*16 + j
        union { v16h v; v8h h[2]; } au;
        au.h[0] = *(const v8h*)(arow + ks * 32);
        au.h[1] = *(const v8h*)(arow + ks * 32 + 8);
        // B: packed fragment, one 32-byte load
        v16h b = bfrag[(long long)ks * NTILES_N * 32];
        acc = __builtin_amdgcn_wmma_f32_16x16x32_f16(
            /*neg_a=*/false, au.v, /*neg_b=*/false, b,
            /*c_mod=*/(short)0, acc, /*reuse_a=*/false, /*reuse_b=*/false);
    }
    // C/D layout: VGPR v holds M = hi*8 + v, N = lane&15
    const int col = nt * 16 + lm;
    const float bv = bias[col];
    float* xout = X + (long long)(m0 + hi * 8) * KD + col;
#pragma unroll
    for (int v = 0; v < 8; ++v) {
        xout[(long long)v * KD] = acc[v] + bv;
    }
}

// ---------------------------------------------------------------------------
// edge helpers: src/dst/type for edge e (self loops appended at e >= EE)
// ---------------------------------------------------------------------------
__device__ __forceinline__ void edge_sdt(const int* __restrict__ ei,
                                         const int* __restrict__ et,
                                         int e, int& s, int& d, int& t) {
    if (e < EE) { s = ei[e]; d = ei[EE + e]; t = et[e]; }
    else        { s = e - EE; d = e - EE;    t = SELF_REL; }
}

// ---------------------------------------------------------------------------
// kernel 2: per-edge attention logits + segment max (one wave32 per edge)
//   alpha[e,k] = leakyrelu( sum_d (x_i*w)(x_j*w) )
// ---------------------------------------------------------------------------
__global__ __launch_bounds__(256)
void k_edge_logits(const float* __restrict__ X,        // [NN,384]
                   const float* __restrict__ relW,     // [101,384]
                   const int* __restrict__ ei, const int* __restrict__ et,
                   float* __restrict__ alpha,          // [EF,3]
                   unsigned* __restrict__ amaxe) {     // [NN,3]
    const int e = blockIdx.x * 8 + (threadIdx.x >> 5);
    if (e >= EF) return;
    const int lane = threadIdx.x & 31;
    int s, d, t;
    edge_sdt(ei, et, e, s, d, t);

    const float4* xi = (const float4*)(X + (long long)d * KD) + lane;
    const float4* xj = (const float4*)(X + (long long)s * KD) + lane;
    const float4* we = (const float4*)(relW + (long long)t * KD) + lane;

    float p[KK];
#pragma unroll
    for (int k = 0; k < KK; ++k) {
        float4 a = xi[k * 32];
        float4 b = xj[k * 32];
        float4 w = we[k * 32];
        p[k] = (a.x * w.x) * (b.x * w.x) + (a.y * w.y) * (b.y * w.y) +
               (a.z * w.z) * (b.z * w.z) + (a.w * w.w) * (b.w * w.w);
    }
#pragma unroll
    for (int k = 0; k < KK; ++k)
#pragma unroll
        for (int off = 16; off > 0; off >>= 1) p[k] += __shfl_xor(p[k], off, 32);

    if (lane == 0) {
#pragma unroll
        for (int k = 0; k < KK; ++k) {
            float v = p[k];
            v = (v > 0.0f) ? v : NEG_SLOPE * v;
            alpha[(long long)e * KK + k] = v;
            atomicMax(&amaxe[(long long)d * KK + k], encOrdered(v));
        }
    }
}

// ---------------------------------------------------------------------------
// kernel 3: a = exp(logit - amax[dst]); segment-sum into asum[dst]
// one thread per (edge, factor)
// ---------------------------------------------------------------------------
__global__ __launch_bounds__(256)
void k_edge_expsum(const int* __restrict__ ei, const int* __restrict__ et,
                   float* __restrict__ alpha, const unsigned* __restrict__ amaxe,
                   float* __restrict__ asum) {
    long long gid = (long long)blockIdx.x * blockDim.x + threadIdx.x;
    if (gid >= (long long)EF * KK) return;
    const int e = (int)(gid / KK);
    const int k = (int)(gid % KK);
    int s, d, t;
    edge_sdt(ei, et, e, s, d, t);
    (void)s; (void)t;
    float m = decOrdered(amaxe[(long long)d * KK + k]);
    float a = __expf(alpha[gid] - m);
    alpha[gid] = a;
    atomicAdd(&asum[(long long)d * KK + k], a);
}

// ---------------------------------------------------------------------------
// kernel 4: aggregate  out[dst] += x_j * w * (1 + r) * alpha_norm
// (cross opn: x_j*r*w + x_j*w == x_j*w*(1+r));  one wave32 per edge
// ---------------------------------------------------------------------------
__global__ __launch_bounds__(256)
void k_aggregate(const float* __restrict__ X,
                 const float* __restrict__ relW,     // [101,384]
                 const float* __restrict__ initRel,  // [100,128]
                 const float* __restrict__ loopRel,  // [1,128]
                 const int* __restrict__ ei, const int* __restrict__ et,
                 const float* __restrict__ alpha, const float* __restrict__ asum,
                 float* __restrict__ outagg) {
    const int e = blockIdx.x * 8 + (threadIdx.x >> 5);
    if (e >= EF) return;
    const int lane = threadIdx.x & 31;
    int s, d, t;
    edge_sdt(ei, et, e, s, d, t);

    const float* relRow = (t < SELF_REL) ? (initRel + (long long)t * DD) : loopRel;
    float4 r = ((const float4*)relRow)[lane];
    r.x += 1.0f; r.y += 1.0f; r.z += 1.0f; r.w += 1.0f;

    const float4* xj = (const float4*)(X + (long long)s * KD) + lane;
    const float4* we = (const float4*)(relW + (long long)t * KD) + lane;
    float* ob = outagg + (long long)d * KD + lane * 4;

#pragma unroll
    for (int k = 0; k < KK; ++k) {
        float an = alpha[(long long)e * KK + k] /
                   (asum[(long long)d * KK + k] + 1e-16f);
        float4 b = xj[k * 32];
        float4 w = we[k * 32];
        float* o = ob + k * DD;
        atomicAdd(o + 0, b.x * w.x * r.x * an);
        atomicAdd(o + 1, b.y * w.y * r.y * an);
        atomicAdd(o + 2, b.z * w.z * r.z * an);
        atomicAdd(o + 3, b.w * w.w * r.w * an);
    }
}

// ---------------------------------------------------------------------------
// kernel 5: BatchNorm column stats (blockDim = 384 => coalesced column reads)
// ---------------------------------------------------------------------------
#define BN_ROWS_PER_BLK 256
__global__ __launch_bounds__(KD)
void k_bn_stats(const float* __restrict__ outagg,
                float* __restrict__ colsum, float* __restrict__ colsumsq) {
    const int c = threadIdx.x;                 // column 0..383
    const int r0 = blockIdx.x * BN_ROWS_PER_BLK;
    const int r1 = min(r0 + BN_ROWS_PER_BLK, NN);
    float s = 0.0f, s2 = 0.0f;
    for (int r = r0; r < r1; ++r) {
        float v = outagg[(long long)r * KD + c];
        s += v; s2 += v * v;
    }
    atomicAdd(&colsum[c], s);
    atomicAdd(&colsumsq[c], s2);
}

// ---------------------------------------------------------------------------
// kernel 6: normalize + affine + tanh -> d_out
// ---------------------------------------------------------------------------
__global__ __launch_bounds__(256)
void k_bn_finalize(const float* __restrict__ outagg,
                   const float* __restrict__ colsum, const float* __restrict__ colsumsq,
                   const float* __restrict__ gamma, const float* __restrict__ beta,
                   float* __restrict__ out) {
    long long gid = (long long)blockIdx.x * blockDim.x + threadIdx.x;
    if (gid >= (long long)NN * KD) return;
    const int c = (int)(gid % KD);
    const float invN = 1.0f / (float)NN;
    float mu = colsum[c] * invN;
    float var = colsumsq[c] * invN - mu * mu;
    float rstd = rsqrtf(var + BN_EPS);
    float v = (outagg[gid] - mu) * rstd * gamma[c] + beta[c];
    out[gid] = tanhf(v);
}

// ---------------------------------------------------------------------------
// launch
// ---------------------------------------------------------------------------
extern "C" void kernel_launch(void* const* d_in, const int* in_sizes, int n_in,
                              void* d_out, int out_size, void* d_ws, size_t ws_size,
                              hipStream_t stream) {
    (void)in_sizes; (void)n_in; (void)out_size; (void)ws_size;

    const float* init_embed = (const float*)d_in[0];   // [NN,128]
    const float* init_rel   = (const float*)d_in[1];   // [100,128]
    const float* pca_w      = (const float*)d_in[2];   // [128,384]
    const float* pca_b      = (const float*)d_in[3];   // [384]
    const float* loop_rel   = (const float*)d_in[4];   // [1,128]
    const float* rel_weight = (const float*)d_in[5];   // [101,3,128]
    const float* bn_gamma   = (const float*)d_in[6];   // [384]
    const float* bn_beta    = (const float*)d_in[7];   // [384]
    const int*   edge_index = (const int*)d_in[8];     // [2,EE]
    const int*   edge_type  = (const int*)d_in[9];     // [EE]
    float*       out        = (float*)d_out;           // [NN,3,128]

    // workspace layout (bytes, 256-aligned)
    char* ws = (char*)d_ws;
    float*     x        = (float*)(ws);                      // 76,800,000
    float*     outagg   = (float*)(ws + 76800000LL);         // 76,800,000
    float*     alpha    = (float*)(ws + 153600000LL);        //  6,600,000
    unsigned*  amaxe    = (unsigned*)(ws + 160200192LL);     //    600,000
    float*     asum     = (float*)(ws + 160800256LL);        //    600,000
    float*     colsum   = (float*)(ws + 161400320LL);        //      1,536
    float*     colsumsq = (float*)(ws + 161401856LL);        //      1,536
    _Float16*  Ah       = (_Float16*)(ws + 161403392LL);     // 12,800,000
    _Float16*  Wp       = (_Float16*)(ws + 174203392LL);     //     98,304

    // 0) init accumulators
    k_init<<<2048, 256, 0, stream>>>(outagg, asum, amaxe, colsum, colsumsq);

    // P1/P2) pack GEMM operands into WMMA-friendly f16 layouts
    {
        long long n4 = (long long)NN * DD / 4;
        k_pack_a<<<(int)((n4 + 255) / 256), 256, 0, stream>>>(init_embed, Ah);
        k_pack_w<<<(DD * KD + 255) / 256, 256, 0, stream>>>(pca_w, Wp);
    }

    // 1) PCA GEMM via WMMA: 3125 M-tiles x 24 N-tiles
    k_pca_wmma<<<(NN / 16) * NTILES_N, 32, 0, stream>>>(Ah, Wp, pca_b, x);

    // 2) attention logits + segment max   (8 edges / 256-thread block)
    k_edge_logits<<<(EF + 7) / 8, 256, 0, stream>>>(x, rel_weight, edge_index,
                                                    edge_type, alpha, amaxe);

    // 3) exp + segment sum
    {
        long long total = (long long)EF * KK;
        k_edge_expsum<<<(int)((total + 255) / 256), 256, 0, stream>>>(
            edge_index, edge_type, alpha, amaxe, asum);
    }

    // 4) weighted message aggregation
    k_aggregate<<<(EF + 7) / 8, 256, 0, stream>>>(x, rel_weight, init_rel, loop_rel,
                                                  edge_index, edge_type, alpha, asum,
                                                  outagg);

    // 5) BN column stats
    k_bn_stats<<<(NN + BN_ROWS_PER_BLK - 1) / BN_ROWS_PER_BLK, KD, 0, stream>>>(
        outagg, colsum, colsumsq);

    // 6) normalize + tanh
    {
        long long total = (long long)NN * KD;
        k_bn_finalize<<<(int)((total + 255) / 256), 256, 0, stream>>>(
            outagg, colsum, colsumsq, bn_gamma, bn_beta, out);
    }
}
